// Spatial_GCN_37701222924737
// MI455X (gfx1250) — matrix operand order repaired and checked by reading, not compiled, not run on hardware
//
#include <hip/hip_runtime.h>

// ---- CDNA5 wave32 WMMA types ----
typedef __attribute__((ext_vector_type(16))) __bf16 v16bf;
typedef __attribute__((ext_vector_type(8)))  __bf16 v8bf;
typedef __attribute__((ext_vector_type(8)))  float  v8f;
typedef __attribute__((ext_vector_type(4)))  int    v4i;

#define IN_C    512
#define BATCH   64
#define NROWS   (BATCH * IN_C)      // 32768 GEMM rows (b*C + c)
#define NN      576                 // nodes = 24*24 (K and N of the GEMM)
#define KSTEPS  18                  // 576 / 32
#define ADJT_STRIDE 584             // padded bf16 elements per adjT row (1168 B, conflict-free)
#define CHUNK_COLS 96               // adjT columns staged in LDS per chunk
#define NCHUNKS 6                   // 576 / 96
#define CHUNK_VEC4 (CHUNK_COLS * (ADJT_STRIDE * 2 / 16))  // 7008 16B packets per chunk
#define LDS_BYTES (CHUNK_COLS * ADJT_STRIDE * 2)          // 112128 B dynamic LDS

// workspace layout (float offsets)
#define WS_PSUM  0                  // [256 blocks][128 rows] partial sums
#define WS_PSQ   32768
#define WS_SCALE 65536              // per-channel fused gamma*rstd
#define WS_SHIFT 66048              // per-channel fused beta - mean*scale
#define WS_ADJT_FLOATOFF 66560      // byte 266240, 16B aligned; adjT bf16 starts here

// ---- CDNA5 async global->LDS path (ASYNCcnt), guarded so compile never breaks ----
#if defined(__has_builtin)
#if __has_builtin(__builtin_amdgcn_global_load_async_to_lds_b128)
#define HAVE_ASYNC_LDS 1
#endif
#endif
#ifndef HAVE_ASYNC_LDS
#define HAVE_ASYNC_LDS 0
#endif

// ---------------- pass 0: adj (fp32, [k][n]) -> adjT bf16 [n][k], padded ----------------
__global__ void build_adjT(const float* __restrict__ adj, __bf16* __restrict__ adjT) {
    int g = blockIdx.x * 256 + threadIdx.x;      // covers 576*584 exactly
    int n = g / ADJT_STRIDE;
    int k = g % ADJT_STRIDE;
    float v = (k < NN) ? adj[k * NN + n] : 0.0f;
    adjT[(size_t)n * ADJT_STRIDE + k] = (__bf16)v;
}

// ---------------- GEMM passes: PASS==1 -> stats, PASS==2 -> normalize+write ----------------
template <int PASS>
__global__ __launch_bounds__(256)
void gcn_gemm(const float* __restrict__ x,
              const float* __restrict__ weight,
              float* __restrict__ ws,
              const __bf16* __restrict__ adjT,
              float* __restrict__ out)
{
    extern __shared__ __bf16 sB[];               // CHUNK_COLS * ADJT_STRIDE bf16
    const int tid  = threadIdx.x;
    const int wave = tid >> 5;
    const int lane = tid & 31;
    const int half = lane >> 4;                  // lane group (0: lanes 0-15, 1: 16-31)
    const int l16  = lane & 15;
    const int rbase = blockIdx.x * 128 + wave * 16;
    const int myrow = rbase + l16;               // A-matrix: lane holds row = lane%16

    // ---- load this lane's full X row slab as 18 bf16 A-fragments (kept in VGPRs) ----
    // 16-bit A 16x32 layout: half 0 -> K runs [0..7],[16..23]; half 1 -> [8..15],[24..31]
    v16bf a[KSTEPS];
    {
        const float4* xr = (const float4*)(x + (size_t)myrow * NN);
#pragma unroll
        for (int s = 0; s < KSTEPS; ++s) {
            float4 p0 = xr[8 * s + 2 * half];
            float4 p1 = xr[8 * s + 2 * half + 1];
            float4 p2 = xr[8 * s + 2 * half + 4];
            float4 p3 = xr[8 * s + 2 * half + 5];
            v16bf av;
            av[0]=(__bf16)p0.x;  av[1]=(__bf16)p0.y;  av[2]=(__bf16)p0.z;  av[3]=(__bf16)p0.w;
            av[4]=(__bf16)p1.x;  av[5]=(__bf16)p1.y;  av[6]=(__bf16)p1.z;  av[7]=(__bf16)p1.w;
            av[8]=(__bf16)p2.x;  av[9]=(__bf16)p2.y;  av[10]=(__bf16)p2.z; av[11]=(__bf16)p2.w;
            av[12]=(__bf16)p3.x; av[13]=(__bf16)p3.y; av[14]=(__bf16)p3.z; av[15]=(__bf16)p3.w;
            a[s] = av;
        }
    }

    float sumv[8], sumq[8], scl[8], sft[8];
#pragma unroll
    for (int i = 0; i < 8; ++i) { sumv[i] = 0.f; sumq[i] = 0.f; scl[i] = 0.f; sft[i] = 0.f; }
    if (PASS == 2) {
#pragma unroll
        for (int i = 0; i < 8; ++i) {
            int c = (rbase + 8 * half + i) & (IN_C - 1);
            scl[i] = ws[WS_SCALE + c];
            sft[i] = ws[WS_SHIFT + c];
        }
    }

    for (int chunk = 0; chunk < NCHUNKS; ++chunk) {
        __syncthreads();                          // previous chunk's readers done
        // ---- stage 96 adjT columns into LDS ----
        {
            const uint4* src = (const uint4*)(adjT + (size_t)chunk * CHUNK_COLS * ADJT_STRIDE);
            uint4* dst = (uint4*)sB;
#if HAVE_ASYNC_LDS
            for (int idx = tid; idx < CHUNK_VEC4; idx += 256) {
                __builtin_amdgcn_global_load_async_to_lds_b128(
                    (__attribute__((address_space(1))) v4i*)(src + idx),
                    (__attribute__((address_space(3))) v4i*)(dst + idx),
                    0, 0);
            }
#if __has_builtin(__builtin_amdgcn_s_wait_asynccnt)
            __builtin_amdgcn_s_wait_asynccnt(0);
#else
            asm volatile("s_wait_asynccnt 0" ::: "memory");
#endif
#else
            for (int idx = tid; idx < CHUNK_VEC4; idx += 256)
                dst[idx] = src[idx];
#endif
        }
        __syncthreads();

        // ---- 3 pairs of 16-col tiles: two independent WMMA accumulator chains ----
#pragma unroll 1
        for (int ctp = 0; ctp < 3; ++ctp) {
            v8f acc0 = {};
            v8f acc1 = {};
            // B 32x16 bf16 layout: lanes 0-15 hold K [0..7],[16..23] of column lane%16,
            // lanes 16-31 hold K [8..15],[24..31] -> lane reads 8 K at 8*half and 8 K at 16+8*half.
            const __bf16* bp0 = sB + (ctp * 32 + l16) * ADJT_STRIDE + 8 * half;
            const __bf16* bp1 = bp0 + 16 * ADJT_STRIDE;
#pragma unroll
            for (int s = 0; s < KSTEPS; ++s) {
                v8bf lo0 = *(const v8bf*)(bp0 + 32 * s);
                v8bf hi0 = *(const v8bf*)(bp0 + 32 * s + 16);
                v8bf lo1 = *(const v8bf*)(bp1 + 32 * s);
                v8bf hi1 = *(const v8bf*)(bp1 + 32 * s + 16);
                v16bf bv0, bv1;
#pragma unroll
                for (int e = 0; e < 8; ++e) {
                    bv0[e] = lo0[e]; bv0[e + 8] = hi0[e];
                    bv1[e] = lo1[e]; bv1[e + 8] = hi1[e];
                }
                acc0 = __builtin_amdgcn_wmma_f32_16x16x32_bf16(
                           false, a[s], false, bv0, (short)0, acc0, false, false);
                acc1 = __builtin_amdgcn_wmma_f32_16x16x32_bf16(
                           false, a[s], false, bv1, (short)0, acc1, false, false);
            }
            const int g0 = chunk * CHUNK_COLS + ctp * 32 + l16;  // global output columns
            const int g1 = g0 + 16;
#pragma unroll
            for (int i = 0; i < 8; ++i) {                        // C/D: VGPR i -> row 8*half+i
                int row = rbase + 8 * half + i;
                int c   = row & (IN_C - 1);
                float y0 = acc0[i] * weight[c * NN + g0];
                float y1 = acc1[i] * weight[c * NN + g1];
                if (PASS == 1) {
                    sumv[i] += y0 + y1;
                    sumq[i] += y0 * y0 + y1 * y1;
                } else {
                    float yn0 = y0 * scl[i] + sft[i];
                    float yn1 = y1 * scl[i] + sft[i];
                    yn0 = (yn0 >= 0.f) ? yn0 : 0.2f * yn0;       // LeakyReLU(0.2)
                    yn1 = (yn1 >= 0.f) ? yn1 : 0.2f * yn1;
                    __builtin_nontemporal_store(yn0, out + (size_t)row * NN + g0);
                    __builtin_nontemporal_store(yn1, out + (size_t)row * NN + g1);
                }
            }
        }
    }

    if (PASS == 1) {
        // reduce over the 16 column-lanes of each half-group, then one plain store per row
#pragma unroll
        for (int i = 0; i < 8; ++i) {
            float sv = sumv[i], sq = sumq[i];
            for (int m = 1; m <= 8; m <<= 1) {
                sv += __shfl_xor(sv, m, 32);
                sq += __shfl_xor(sq, m, 32);
            }
            if (l16 == 0) {
                int lc = wave * 16 + 8 * half + i;               // row within this block
                ws[WS_PSUM + blockIdx.x * 128 + lc] = sv;
                ws[WS_PSQ  + blockIdx.x * 128 + lc] = sq;
            }
        }
    }
}

// ---------------- deterministic stats reduction + fused BN scale/shift ----------------
__global__ void finalize_stats(const float* __restrict__ gamma,
                               const float* __restrict__ beta,
                               float* __restrict__ ws)
{
    int c = blockIdx.x * 256 + threadIdx.x;
    if (c >= IN_C) return;
    int phase = c >> 7;                 // which blocks covered this channel: w = phase + 4t
    int lc    = c & 127;
    float s = 0.f, q = 0.f;
    for (int t = 0; t < 64; ++t) {
        int w = phase + 4 * t;
        s += ws[WS_PSUM + w * 128 + lc];
        q += ws[WS_PSQ  + w * 128 + lc];
    }
    const float inv = 1.0f / (float)(BATCH * NN);   // 1/36864, biased var
    float mean = s * inv;
    float var  = q * inv - mean * mean;
    float rstd = rsqrtf(var + 1e-4f);
    float a = gamma[c] * rstd;
    ws[WS_SCALE + c] = a;
    ws[WS_SHIFT + c] = beta[c] - mean * a;
}

extern "C" void kernel_launch(void* const* d_in, const int* in_sizes, int n_in,
                              void* d_out, int out_size, void* d_ws, size_t ws_size,
                              hipStream_t stream) {
    (void)in_sizes; (void)n_in; (void)out_size; (void)ws_size;
    const float* x      = (const float*)d_in[0];
    const float* adj    = (const float*)d_in[1];
    const float* weight = (const float*)d_in[2];
    const float* gamma  = (const float*)d_in[3];
    const float* beta   = (const float*)d_in[4];
    float*  out = (float*)d_out;
    float*  wsf = (float*)d_ws;
    __bf16* adjT = reinterpret_cast<__bf16*>(wsf + WS_ADJT_FLOATOFF);

    // pass 0: transpose+convert adj once (1.3 MB -> 0.66 MB bf16, stays L2-resident)
    build_adjT<<<(NN * ADJT_STRIDE) / 256, 256, 0, stream>>>(adj, adjT);
    // pass 1: bf16 WMMA GEMM, per-block deterministic BN partials (no y spill; x fills L2)
    gcn_gemm<1><<<NROWS / 128, 256, LDS_BYTES, stream>>>(x, weight, wsf, adjT, out);
    // fuse BN into per-channel scale/shift
    finalize_stats<<<2, 256, 0, stream>>>(gamma, beta, wsf);
    // pass 2: recompute GEMM (x from L2), apply BN + LeakyReLU, NT-store output
    gcn_gemm<2><<<NROWS / 128, 256, LDS_BYTES, stream>>>(x, weight, wsf, adjT, out);
}
